// CrossAttention_2d_43611097923689
// MI455X (gfx1250) — matrix operand order repaired and checked by reading, not compiled, not run on hardware
//
#include <hip/hip_runtime.h>
#include <math.h>

typedef __attribute__((ext_vector_type(16))) _Float16 v16h;
typedef __attribute__((ext_vector_type(8)))  _Float16 v8h;
typedef __attribute__((ext_vector_type(4)))  _Float16 v4h;
typedef __attribute__((ext_vector_type(8)))  float    v8f;
typedef __attribute__((ext_vector_type(4)))  float    v4f;
typedef __attribute__((ext_vector_type(4)))  unsigned int v4u;
typedef __attribute__((ext_vector_type(8)))  int      v8i;
typedef __attribute__((ext_vector_type(4)))  int      v4i;

#define D_DIM    512
#define S_DIM    1281
#define BM       64          // query rows per workgroup
#define BN       64          // key rows per iteration
#define NTHREADS 128
#define KROW_LD  520         // halves: 512 + 8 pad (bank-conflict break)
#define KCOL_LD  72          // halves: 64 + 8 pad
#define P_LD     72
#define SCALE    0.044194173824159216f   // 1/sqrt(512)

static __device__ __forceinline__ v16h cat8(v8h lo, v8h hi) {
  return __builtin_shufflevector(lo, hi, 0,1,2,3,4,5,6,7,8,9,10,11,12,13,14,15);
}

// Issue a TDM 2-D tile load: rows x 512 f32 from gsrc (row stride 512) -> LDS.
// Rows beyond rows_in_tensor are zero-filled by TDM OOB handling (ragged tail).
// Descriptor packing per CDNA5 ISA 8.3/8.4 (group0 128b, group1 256b).
static __device__ __forceinline__ void tdm_load_tile(const float* gsrc,
                                                     unsigned lds_off,
                                                     int rows_in_tensor) {
  const unsigned long long ga = (unsigned long long)(size_t)gsrc;
  v4u g0;
  g0[0] = 1u;                                   // count=1, user descriptor
  g0[1] = lds_off;                              // lds_addr (bytes)
  g0[2] = (unsigned)(ga & 0xFFFFFFFFu);         // global_addr[31:0]
  g0[3] = (unsigned)((ga >> 32) & 0x01FFFFFFu)  // global_addr[56:32]
        | (2u << 30);                           // type = 2 ("image")
  v8i g1;
  g1[0] = (int)(2u << 16);                      // wg_mask=0 | data_size=2 (4B)
  g1[1] = (int)((unsigned)D_DIM << 16);         // abar_addr=0 | tensor_dim0.lo16
  g1[2] = (int)(((unsigned)rows_in_tensor & 0xFFFFu) << 16); // dim0.hi16=0 | tensor_dim1.lo16
  g1[3] = (int)(((unsigned)rows_in_tensor >> 16) & 0xFFFFu)  // tensor_dim1.hi16
        | (int)((unsigned)D_DIM << 16);         // tile_dim0 = 512
  g1[4] = BN;                                   // tile_dim1 = 64 | tile_dim2 = 0
  g1[5] = D_DIM;                                // tensor_dim0_stride.lo32 = 512
  g1[6] = 0;                                    // stride0.hi16 | stride1.lo16
  g1[7] = 0;                                    // stride1.hi32
  const v4i z4 = (v4i){0, 0, 0, 0};             // groups 2/3 unused for 2-D
  const v8i z8 = (v8i){0, 0, 0, 0, 0, 0, 0, 0}; // extra group (clang-23 form) unused
  __builtin_amdgcn_tensor_load_to_lds(g0, g1, z4, z4, z8, 0);
}

// Flash cross-attention: out[b,s,:] (+)= softmax_t(Q[b,s,:]·K[b,t,:] * SCALE) @ K
// Tensors are raw-reshaped (B,S,D) row-major views of the (B,D,H,W) buffers.
__global__ __launch_bounds__(NTHREADS, 1)
void flash_xattn(const float* __restrict__ Qg, const float* __restrict__ Kg,
                 float* __restrict__ Og, int accumulate)
{
  __shared__ float    sF32[BN * D_DIM];        // TDM target: raw f32 K tile (128 KB)
  __shared__ _Float16 sKrow[BN * KROW_LD];     // [t_local][d]   : feeds QK^T B-frags
  __shared__ _Float16 sKcol[D_DIM * KCOL_LD];  // [d][t_local]   : feeds P*V  B-frags
  __shared__ _Float16 sP[BM * P_LD];           // [m_local][t]   : softmaxed scores (f16)

  const int tid  = threadIdx.x;
  const int lane = tid & 31;
  const int w    = tid >> 5;          // wave 0..3 -> query rows [16w,16w+16)
  const int g    = (lane >> 4) & 1;   // lane half-group
  const int ln   = lane & 15;
  const float NEG_INF = -__builtin_inff();

  const int b  = blockIdx.y;
  const int q0 = blockIdx.x * BM;

  const float* __restrict__ Qb = Qg + (size_t)b * S_DIM * D_DIM;
  const float* __restrict__ Kb = Kg + (size_t)b * S_DIM * D_DIM;
  float*       __restrict__ Ob = Og + (size_t)b * S_DIM * D_DIM;

  const unsigned sF32_off = (unsigned)(size_t)(void*)&sF32[0];

  // Kick off DMA of key tile 0 while we set up Q fragments.
  if (w == 0)
    tdm_load_tile(Kb, sF32_off, S_DIM);

  // ---- Load Q A-fragments once, pre-scaled by 1/sqrt(D) ----
  // 16-bit A 16x32 layout: lanes<16: e0..7 -> k 0..7, e8..15 -> k 16..23;
  //                        lanes>=16: e0..7 -> k 8..15, e8..15 -> k 24..31.
  v16h qfrag[16];
  {
    int srow = q0 + 16 * w + ln;
    if (srow >= S_DIM) srow = S_DIM - 1;   // ragged tail: clamp (masked at store)
    const float* qp = Qb + (size_t)srow * D_DIM;
    #pragma unroll
    for (int j = 0; j < 16; ++j) {
      const int dA = 32 * j + 8 * g;         // elements 0..7
      const int dB = 32 * j + 16 + 8 * g;    // elements 8..15
      v4f a0 = *(const v4f*)(qp + dA);
      v4f a1 = *(const v4f*)(qp + dA + 4);
      v4f b0 = *(const v4f*)(qp + dB);
      v4f b1 = *(const v4f*)(qp + dB + 4);
      v16h q;
      #pragma unroll
      for (int e = 0; e < 4; ++e) {
        q[e]      = (_Float16)(a0[e] * SCALE);
        q[4 + e]  = (_Float16)(a1[e] * SCALE);
        q[8 + e]  = (_Float16)(b0[e] * SCALE);
        q[12 + e] = (_Float16)(b1[e] * SCALE);
      }
      qfrag[j] = q;
    }
  }

  v8f acc[32];                     // 16 rows x 512 d, C-frag layout per 16x16 tile
  #pragma unroll
  for (int i = 0; i < 32; ++i) acc[i] = (v8f){0.f,0.f,0.f,0.f,0.f,0.f,0.f,0.f};
  float mi[8], li[8];
  #pragma unroll
  for (int r = 0; r < 8; ++r) { mi[r] = NEG_INF; li[r] = 0.f; }

  const int d4 = tid * 4;          // this thread's d-chunk for staging (128 chunks)

  for (int t0 = 0; t0 < S_DIM; t0 += BN) {
    // ---- Wait for DMA of this tile; make it visible to all waves ----
    if (w == 0)
      __builtin_amdgcn_s_wait_tensorcnt(0);
    __syncthreads();

    // ---- Convert f32 tile -> f16 in both layouts (4x4 micro-blocks) ----
    #pragma unroll
    for (int rg = 0; rg < 16; ++rg) {
      const int tl = rg * 4;
      v4h h[4];
      #pragma unroll
      for (int i = 0; i < 4; ++i) {
        v4f v = *(const v4f*)(&sF32[(tl + i) * D_DIM + d4]);
        h[i] = (v4h){ (_Float16)v[0], (_Float16)v[1], (_Float16)v[2], (_Float16)v[3] };
        *(v4h*)(&sKrow[(tl + i) * KROW_LD + d4]) = h[i];
      }
      #pragma unroll
      for (int e = 0; e < 4; ++e) {
        v4h c = { h[0][e], h[1][e], h[2][e], h[3][e] };
        *(v4h*)(&sKcol[(d4 + e) * KCOL_LD + tl]) = c;   // 4 t's packed per d
      }
    }
    __syncthreads();   // f32 buffer free, f16 tiles ready

    // ---- Start DMA of the next tile; overlaps all compute below ----
    if (w == 0 && t0 + BN < S_DIM)
      tdm_load_tile(Kb + (size_t)(t0 + BN) * D_DIM, sF32_off, S_DIM - (t0 + BN));

    // ---- Scores: 16 q-rows x 64 keys, 64 WMMAs, B-frags pipelined 2 ahead ----
    v8f sc[4];
    #pragma unroll
    for (int n = 0; n < 4; ++n) sc[n] = (v8f){0.f,0.f,0.f,0.f,0.f,0.f,0.f,0.f};
    {
      auto loadB = [&](int idx) -> v16h {
        const int j = idx >> 2, n = idx & 3;
        // B = K^T: lane col t = 16n+ln, k(d) = 32j + 8g + e (e<8), 32j + 16 + 8g + (e-8) (e>=8)
        const _Float16* kp = &sKrow[(16 * n + ln) * KROW_LD + 32 * j + 8 * g];
        return cat8(*(const v8h*)kp, *(const v8h*)(kp + 16));
      };
      v16h cur = loadB(0);
      v16h nxt = loadB(1);
      #pragma unroll
      for (int idx = 0; idx < 64; ++idx) {
        const int j = idx >> 2, n = idx & 3;
        v16h fut = (idx + 2 < 64) ? loadB(idx + 2) : cur;
        sc[n] = __builtin_amdgcn_wmma_f32_16x16x32_f16(
            false, qfrag[j], false, cur, (short)0, sc[n], false, false);
        cur = nxt;
        nxt = fut;
      }
    }

    // ---- Mask invalid key columns (scale already folded into Q) ----
    #pragma unroll
    for (int n = 0; n < 4; ++n) {
      const bool valid = (t0 + 16 * n + ln) < S_DIM;
      #pragma unroll
      for (int r = 0; r < 8; ++r)
        sc[n][r] = valid ? sc[n][r] : NEG_INF;
    }

    // ---- Online softmax (row stats across 16 lanes of each half-group) ----
    float corr[8];
    #pragma unroll
    for (int r = 0; r < 8; ++r) {
      float m = sc[0][r];
      m = fmaxf(m, sc[1][r]); m = fmaxf(m, sc[2][r]); m = fmaxf(m, sc[3][r]);
      m = fmaxf(m, __shfl_xor(m, 1, 32));
      m = fmaxf(m, __shfl_xor(m, 2, 32));
      m = fmaxf(m, __shfl_xor(m, 4, 32));
      m = fmaxf(m, __shfl_xor(m, 8, 32));
      const float mnew = fmaxf(mi[r], m);
      const float c = __expf(mi[r] - mnew);
      mi[r] = mnew;
      float rs = 0.f;
      #pragma unroll
      for (int n = 0; n < 4; ++n) {
        float p = __expf(sc[n][r] - mnew);
        sc[n][r] = p;
        rs += p;
      }
      rs += __shfl_xor(rs, 1, 32);
      rs += __shfl_xor(rs, 2, 32);
      rs += __shfl_xor(rs, 4, 32);
      rs += __shfl_xor(rs, 8, 32);
      li[r] = li[r] * c + rs;
      corr[r] = c;
    }

    // ---- Rescale accumulator, stash P (f16) in this wave's LDS strip ----
    #pragma unroll
    for (int dt = 0; dt < 32; ++dt) {
      #pragma unroll
      for (int r = 0; r < 8; ++r) acc[dt][r] *= corr[r];
    }
    #pragma unroll
    for (int n = 0; n < 4; ++n) {
      #pragma unroll
      for (int r = 0; r < 8; ++r)
        sP[(16 * w + r + 8 * g) * P_LD + 16 * n + ln] = (_Float16)sc[n][r];
    }

    // ---- P (16x64) @ V (64x512): 64 WMMAs, B-frags pipelined 2 ahead ----
    {
      v16h pfrag[2];
      #pragma unroll
      for (int kk = 0; kk < 2; ++kk) {
        const _Float16* pp = &sP[(16 * w + ln) * P_LD + 32 * kk + 8 * g];
        pfrag[kk] = cat8(*(const v8h*)pp, *(const v8h*)(pp + 16));
      }
      auto loadV = [&](int idx) -> v16h {
        const int dt = idx >> 1, kk = idx & 1;
        // B = V: lane col d = 16dt+ln, k(t) = 32kk + 8g + e (e<8), 32kk + 16 + 8g + (e-8) (e>=8)
        const _Float16* vp = &sKcol[(16 * dt + ln) * KCOL_LD + 32 * kk + 8 * g];
        return cat8(*(const v8h*)vp, *(const v8h*)(vp + 16));
      };
      v16h cur = loadV(0);
      v16h nxt = loadV(1);
      #pragma unroll
      for (int idx = 0; idx < 64; ++idx) {
        const int dt = idx >> 1, kk = idx & 1;
        v16h fut = (idx + 2 < 64) ? loadV(idx + 2) : cur;
        acc[dt] = __builtin_amdgcn_wmma_f32_16x16x32_f16(
            false, pfrag[kk], false, cur, (short)0, acc[dt], false, false);
        cur = nxt;
        nxt = fut;
      }
    }
    __syncthreads();   // all waves done with sKrow/sKcol before next conversion
  }

  // ---- Epilogue: normalize and write (pass 2 accumulates) ----
  #pragma unroll
  for (int r = 0; r < 8; ++r) {
    const int srow = q0 + 16 * w + r + 8 * g;
    if (srow >= S_DIM) continue;
    const float inv = 1.f / li[r];
    float* op = Ob + (size_t)srow * D_DIM;
    #pragma unroll
    for (int dt = 0; dt < 32; ++dt) {
      const int d = 16 * dt + ln;
      const float v = acc[dt][r] * inv;
      if (accumulate) op[d] += v;
      else            op[d] = v;
    }
  }
}

extern "C" void kernel_launch(void* const* d_in, const int* in_sizes, int n_in,
                              void* d_out, int out_size, void* d_ws, size_t ws_size,
                              hipStream_t stream) {
  (void)n_in; (void)out_size; (void)d_ws; (void)ws_size;
  const float* lidar  = (const float*)d_in[0];
  const float* visual = (const float*)d_in[1];
  float* out = (float*)d_out;

  const int B = in_sizes[0] / (S_DIM * D_DIM);   // 32
  dim3 grid((S_DIM + BM - 1) / BM, B);           // (21, 32)
  dim3 block(NTHREADS);

  // out1: lidar queries attend to visual K/V (overwrites d_out)
  flash_xattn<<<grid, block, 0, stream>>>(lidar, visual, out, 0);
  // out2: visual queries attend to lidar K/V (accumulates; stream-ordered)
  flash_xattn<<<grid, block, 0, stream>>>(visual, lidar, out, 1);
}
